// cell_40475771798012
// MI455X (gfx1250) — matrix operand order repaired and checked by reading, not compiled, not run on hardware
//
#include <hip/hip_runtime.h>
#include <hip/hip_bf16.h>
#include <math.h>

// ---------------------------------------------------------------------------
// CDNA5 (gfx1250) implementation.
//  Stage 1: pointwise MLPs (VALU, scalar weights)        -> Qh/Kh/Vh (f16, ws)
//  Stage 2: circular cross-corr via Toeplitz-A f16 WMMA  (v_wmma_f32_16x16x32_f16)
//           fused with top-7 / softmax / delay-aggregation (corr lives in LDS only)
//  Stage 3: adj@adj + spatial attention + GCN via fp32 WMMA (v_wmma_f32_16x16x4_f32)
// ---------------------------------------------------------------------------

typedef __attribute__((ext_vector_type(16))) _Float16 v16h;
typedef __attribute__((ext_vector_type(8)))  float    v8f;
typedef __attribute__((ext_vector_type(2)))  float    v2f;

#define NB  207
#define BB  16
#define TS  288
#define TL  2016
#define MM  4
#define PL  12
#define TK  7
#define BN  (BB*NB)        /* 3312  */
#define ROWS (MM*BN)       /* 13248 */

// d_out layout (floats): out | delay | tc | A
#define OUT_OFF   0
#define DELAY_OFF (MM*BB*NB*PL)              /* 158976 */
#define TC_OFF    (DELAY_OFF + MM*BB*NB*TK)  /* 251712 */
#define A_OFF     (TC_OFF + MM*BB*NB*TK)     /* 344448 */

__device__ __forceinline__ v8f wmma_f32_4(v2f a, v2f b, v8f c) {
  return __builtin_amdgcn_wmma_f32_16x16x4_f32(false, a, false, b, (short)0, c, false, false);
}

// ------------------------- Stage 1: pointwise MLPs -------------------------

__global__ __launch_bounds__(256) void k_mlp_recent(
    const float* __restrict__ x_in,
    const float* __restrict__ W1, const float* __restrict__ b1,
    const float* __restrict__ W2, const float* __restrict__ b2,
    _Float16* __restrict__ Qh)
{
  int idx = blockIdx.x * 256 + threadIdx.x;
  if (idx >= BN * TS) return;
  int t = idx % TS;
  int bn = idx / TS;
  float x = x_in[idx];
  float a0 = b2[0], a1 = b2[1], a2 = b2[2], a3 = b2[3];
  for (int j = 0; j < 64; j++) {
    float h = fmaxf(x * W1[j] + b1[j], 0.f);
    a0 += h * W2[j*4+0]; a1 += h * W2[j*4+1];
    a2 += h * W2[j*4+2]; a3 += h * W2[j*4+3];
  }
  float a[4] = {a0, a1, a2, a3};
  #pragma unroll
  for (int m = 0; m < 4; m++)
    Qh[(size_t)(m*BN + bn)*TS + t] = (_Float16)a[m];
}

__global__ __launch_bounds__(256) void k_mlp_hist(
    const float* __restrict__ x_in,
    const float* __restrict__ Wk1, const float* __restrict__ bk1,
    const float* __restrict__ Wk2, const float* __restrict__ bk2,
    const float* __restrict__ Wv1, const float* __restrict__ bv1,
    const float* __restrict__ Wv2, const float* __restrict__ bv2,
    _Float16* __restrict__ Kh, _Float16* __restrict__ Vh)
{
  int idx = blockIdx.x * 256 + threadIdx.x;
  if (idx >= BN * TL) return;
  int t = idx % TL;
  int bn = idx / TL;
  float x = x_in[idx];
  float k0 = bk2[0], k1 = bk2[1], k2 = bk2[2], k3 = bk2[3];
  float v0 = bv2[0], v1 = bv2[1], v2 = bv2[2], v3 = bv2[3];
  for (int j = 0; j < 64; j++) {
    float hk = fmaxf(x * Wk1[j] + bk1[j], 0.f);
    float hv = fmaxf(x * Wv1[j] + bv1[j], 0.f);
    k0 += hk * Wk2[j*4+0]; k1 += hk * Wk2[j*4+1];
    k2 += hk * Wk2[j*4+2]; k3 += hk * Wk2[j*4+3];
    v0 += hv * Wv2[j*4+0]; v1 += hv * Wv2[j*4+1];
    v2 += hv * Wv2[j*4+2]; v3 += hv * Wv2[j*4+3];
  }
  float ka[4] = {k0, k1, k2, k3};
  float va[4] = {v0, v1, v2, v3};
  #pragma unroll
  for (int m = 0; m < 4; m++) {
    size_t o = (size_t)(m*BN + bn)*TL + t;
    Kh[o] = (_Float16)ka[m];
    Vh[o] = (_Float16)va[m];
  }
}

// ------------- Stage 2: WMMA cross-correlation + topk + aggregate ----------
// corr[tau] = sum_s Q[s] * K[(s - tau) mod 2016]  (FFT-corr identity)
// C[i][j] = corr[tau0 + i + 16j]:
//   A[i][k] = Kext[32c - tau0 + k - i]  (Toeplitz, shared over j)
//   B[k][j] = Qext[32c + 16j + k]       (index shift cancels the 16j delay)

__global__ __launch_bounds__(128) void k_corr(
    const _Float16* __restrict__ Qh, const _Float16* __restrict__ Kh,
    const _Float16* __restrict__ Vh,
    const float* __restrict__ corr_w, const float* __restrict__ corr_b,
    float* __restrict__ delay_out, float* __restrict__ tc_out,
    float* __restrict__ aggW)
{
  __shared__ float Kext[2368];                                   // f32, t in [-2063, 287]
  __shared__ __attribute__((aligned(32))) _Float16 Qext[784];    // f16, s in [-256, 527]
  __shared__ float corrL[TL];
  __shared__ float redV[128];
  __shared__ int   redI[128];
  __shared__ float wtop[TK];
  __shared__ int   dtop[TK];
  __shared__ float tcl[TK];

  const int tid = threadIdx.x;
  const int r = blockIdx.x;
  const _Float16* Krow = Kh + (size_t)r * TL;
  const _Float16* Qrow = Qh + (size_t)r * TS;

  for (int i = tid; i < 2351; i += 128) {
    int t = i - 2063;
    int s = t % TL; if (s < 0) s += TL;
    Kext[i] = (float)Krow[s];
  }
  for (int i = tid; i < 784; i += 128) {
    int s = i - 256;
    Qext[i] = (s >= 0 && s < TS) ? Qrow[s] : (_Float16)0.f;
  }
  __syncthreads();

  const float cw = corr_w[0], cb = corr_b[0];
  const int wv   = tid >> 5;
  const int lane = tid & 31;
  const int lh   = lane >> 4;     // lane half
  const int mr   = lane & 15;
  const int aK0  = lh ? 16 : 0;   // f16 A-layout K split
  const int bK0  = lh ? 16 : 0;   // f16 B-layout K split

  for (int g = wv; g < 8; g += 4) {        // 8 tau-groups of 256, 4 waves
    const int tau0 = g << 8;
    v8f acc = {};
    for (int c = -8; c <= 8; c++) {        // 17 K-chunks of 32
      const int sbase = c << 5;
      v16h af;
      const int tb = 2063 + sbase - tau0 + aK0 - mr;
      #pragma unroll
      for (int e = 0; e < 8; e++) af[e] = (_Float16)Kext[tb + e];
      #pragma unroll
      for (int e = 0; e < 8; e++) af[8 + e] = (_Float16)Kext[tb + 8 + e];
      const v16h bf = *(const v16h*)&Qext[256 + sbase + (mr << 4) + bK0];
      acc = __builtin_amdgcn_wmma_f32_16x16x32_f16(false, af, false, bf,
                                                   (short)0, acc, false, false);
    }
    #pragma unroll
    for (int v = 0; v < 8; v++) {
      int tau = tau0 + (v + (lh << 3)) + (mr << 4);
      if (tau < TL) corrL[tau] = acc[v] * cw + cb;
    }
  }
  __syncthreads();

  // top-7 (descending, lowest-index tie-break) on LDS corr
  for (int kk = 0; kk < TK; kk++) {
    float bv = -3.0e38f; int bi = 0x7fffffff;
    for (int t = tid; t < TL; t += 128) {
      float v = corrL[t];
      if (v > bv || (v == bv && t < bi)) { bv = v; bi = t; }
    }
    redV[tid] = bv; redI[tid] = bi;
    __syncthreads();
    for (int s = 64; s > 0; s >>= 1) {
      if (tid < s) {
        float ov = redV[tid + s]; int oi = redI[tid + s];
        if (ov > redV[tid] || (ov == redV[tid] && oi < redI[tid])) {
          redV[tid] = ov; redI[tid] = oi;
        }
      }
      __syncthreads();
    }
    if (tid == 0) { wtop[kk] = redV[0]; dtop[kk] = redI[0]; corrL[redI[0]] = -3.0e38f; }
    __syncthreads();
  }
  if (tid == 0) {
    float mx = wtop[0], s = 0.f, e[TK];
    for (int k = 0; k < TK; k++) { e[k] = expf(wtop[k] - mx); s += e[k]; }
    for (int k = 0; k < TK; k++) tcl[k] = e[k] / s;
  }
  __syncthreads();
  if (tid < TK) {
    delay_out[(size_t)r * TK + tid] = (float)dtop[tid];
    tc_out[(size_t)r * TK + tid]    = tcl[tid];
  }
  if (tid < 16) {            // time-delay aggregation -> agg (padded to 16 cols)
    float a = 0.f;
    if (tid < PL) {
      const _Float16* Vrow = Vh + (size_t)r * TL;
      for (int k = 0; k < TK; k++) {
        int ix = dtop[k] + PL + tid;
        float vv = (ix < TL) ? (float)Vrow[ix] : 0.f;
        a += tcl[k] * vv;
      }
    }
    aggW[(size_t)r * 16 + tid] = a;
  }
}

// -------------------- Stage 3a: adj @ adj (fp32 WMMA) ----------------------

__global__ __launch_bounds__(32) void k_adjsq(const float* __restrict__ adj,
                                              float* __restrict__ adjsq)
{
  const int bid = blockIdx.x;
  const int it = bid / 13, jt = bid % 13;
  const int lane = threadIdx.x & 31, lh = lane >> 4, mr = lane & 15;
  const int i = it*16 + mr;
  const int j = jt*16 + mr;
  v8f c = {};
  for (int kt = 0; kt < 13; kt++) {
    #pragma unroll
    for (int kk = 0; kk < 16; kk += 4) {
      const int k = kt*16 + kk + (lh << 1);
      v2f a, b;
      a[0] = (i < NB && k     < NB) ? adj[i*NB + k]     : 0.f;
      a[1] = (i < NB && k + 1 < NB) ? adj[i*NB + k + 1] : 0.f;
      b[0] = (k     < NB && j < NB) ? adj[k*NB + j]     : 0.f;
      b[1] = (k + 1 < NB && j < NB) ? adj[(k+1)*NB + j] : 0.f;
      c = wmma_f32_4(a, b, c);
    }
  }
  #pragma unroll
  for (int v = 0; v < 8; v++) {
    const int ii = it*16 + v + (lh << 3);
    if (ii < NB && j < NB) adjsq[ii*NB + j] = c[v];
  }
}

// ------------- Stage 3b: spatial attention + fused GCN (fp32 WMMA) ---------

__global__ __launch_bounds__(256) void k_gcn(
    const float* __restrict__ aggW, const float* __restrict__ adj,
    const float* __restrict__ adjsq,
    const float* __restrict__ spQw, const float* __restrict__ spQb,
    const float* __restrict__ spKw, const float* __restrict__ spKb,
    const float* __restrict__ spW,  const float* __restrict__ gcnW,
    float* __restrict__ outp, float* __restrict__ Aout)
{
  __shared__ float sQW[208*16];   // Qa @ spW  (rows padded to 208, cols to 16)
  __shared__ float sKA[208*16];   // Ka
  __shared__ float sAG[208*16];   // agg
  __shared__ float sA2[8][256];   // per-wave staging: A .* adj
  __shared__ float sA3[8][256];   // per-wave staging: A .* adj^2

  const int tid = threadIdx.x;
  const int mb  = blockIdx.x;

  for (int n = tid; n < 208; n += 256) {
    if (n < NB) {
      float ag[PL], qa[PL], ka[PL];
      const float* arow = aggW + (size_t)(mb*NB + n)*16;
      for (int q = 0; q < PL; q++) ag[q] = arow[q];
      for (int p = 0; p < PL; p++) {
        float s1 = spQb[p], s2 = spKb[p];
        for (int q = 0; q < PL; q++) { s1 += ag[q]*spQw[q*PL+p]; s2 += ag[q]*spKw[q*PL+p]; }
        qa[p] = fmaxf(s1, 0.f); ka[p] = fmaxf(s2, 0.f);
      }
      for (int p = 0; p < PL; p++) {
        float s = 0.f;
        for (int q = 0; q < PL; q++) s += qa[q]*spW[q*PL+p];
        sQW[n*16+p] = s;  sKA[n*16+p] = ka[p];  sAG[n*16+p] = ag[p];
      }
      for (int p = PL; p < 16; p++) { sQW[n*16+p]=0.f; sKA[n*16+p]=0.f; sAG[n*16+p]=0.f; }
    } else {
      for (int p = 0; p < 16; p++) { sQW[n*16+p]=0.f; sKA[n*16+p]=0.f; sAG[n*16+p]=0.f; }
    }
  }
  __syncthreads();

  const int wv = tid >> 5, lane = tid & 31, lh = lane >> 4, mr = lane & 15;
  float* stg2 = sA2[wv];
  float* stg3 = sA3[wv];
  const float g0 = gcnW[0], g1 = gcnW[1], g2 = gcnW[2];

  for (int it = wv; it < 13; it += 8) {
    const int i0 = it << 4;
    v8f cx2 = {}, cx3 = {};
    for (int jt = 0; jt < 13; jt++) {
      const int j0 = jt << 4;
      v8f ca = {};
      // A[i][j] = sum_p QW[i][p] * KA[j][p]   (K = 12 padded to 16)
      #pragma unroll
      for (int kk = 0; kk < 16; kk += 4) {
        const int kb = kk + (lh << 1);
        v2f a, b;
        a[0] = sQW[(i0+mr)*16 + kb]; a[1] = sQW[(i0+mr)*16 + kb + 1];
        b[0] = sKA[(j0+mr)*16 + kb]; b[1] = sKA[(j0+mr)*16 + kb + 1];
        ca = wmma_f32_4(a, b, ca);
      }
      const int j = j0 + mr;
      __builtin_amdgcn_wave_barrier();   // WAR fence vs previous staging reads
      #pragma unroll
      for (int v = 0; v < 8; v++) {
        const int i = i0 + v + (lh << 3);
        float aval = ca[v];
        float a2 = 0.f, a3 = 0.f;
        if (i < NB && j < NB) {
          Aout[((size_t)mb*NB + i)*NB + j] = aval;
          a2 = aval * adj[i*NB + j];
          a3 = aval * adjsq[i*NB + j];
        }
        stg2[(v + (lh<<3))*16 + mr] = a2;
        stg3[(v + (lh<<3))*16 + mr] = a3;
      }
      __builtin_amdgcn_wave_barrier();   // RAW fence: DS in-order within wave
      // x2 += adj2_tile @ agg_tile ; x3 += adj3_tile @ agg_tile
      #pragma unroll
      for (int kk = 0; kk < 16; kk += 4) {
        const int kb = kk + (lh << 1);
        v2f b, a2f, a3f;
        b[0]   = sAG[(j0+kb)*16 + mr]; b[1]   = sAG[(j0+kb+1)*16 + mr];
        a2f[0] = stg2[mr*16 + kb];     a2f[1] = stg2[mr*16 + kb + 1];
        a3f[0] = stg3[mr*16 + kb];     a3f[1] = stg3[mr*16 + kb + 1];
        cx2 = wmma_f32_4(a2f, b, cx2);
        cx3 = wmma_f32_4(a3f, b, cx3);
      }
    }
    #pragma unroll
    for (int v = 0; v < 8; v++) {
      const int i = i0 + v + (lh << 3);
      if (i < NB && mr < PL) {
        float o = g0*sAG[i*16+mr] + g1*fmaxf(cx2[v],0.f) + g2*fmaxf(cx3[v],0.f);
        outp[((size_t)mb*NB + i)*PL + mr] = o;
      }
    }
  }
}

// ------------------------------- launcher ----------------------------------

extern "C" void kernel_launch(void* const* d_in, const int* in_sizes, int n_in,
                              void* d_out, int out_size, void* d_ws, size_t ws_size,
                              hipStream_t stream) {
  (void)in_sizes; (void)n_in; (void)out_size; (void)ws_size;
  const float* recent = (const float*)d_in[0];
  const float* hist   = (const float*)d_in[1];
  const float* Wq1 = (const float*)d_in[2];  const float* bq1 = (const float*)d_in[3];
  const float* Wq2 = (const float*)d_in[4];  const float* bq2 = (const float*)d_in[5];
  const float* Wk1 = (const float*)d_in[6];  const float* bk1 = (const float*)d_in[7];
  const float* Wk2 = (const float*)d_in[8];  const float* bk2 = (const float*)d_in[9];
  const float* Wv1 = (const float*)d_in[10]; const float* bv1 = (const float*)d_in[11];
  const float* Wv2 = (const float*)d_in[12]; const float* bv2 = (const float*)d_in[13];
  const float* corr_w = (const float*)d_in[14];
  const float* corr_b = (const float*)d_in[15];
  const float* gcnW = (const float*)d_in[16];
  const float* adj  = (const float*)d_in[17];
  const float* spQw = (const float*)d_in[18]; const float* spQb = (const float*)d_in[19];
  const float* spKw = (const float*)d_in[20]; const float* spKb = (const float*)d_in[21];
  const float* spW  = (const float*)d_in[22];
  float* out = (float*)d_out;

  // workspace layout (~110 MB): Qh f16 | Kh f16 | Vh f16 | agg f32 | adj^2 f32
  char* ws = (char*)d_ws;
  _Float16* Qh = (_Float16*)ws;                                      //  7,630,848 B
  _Float16* Kh = (_Float16*)(ws + 7630848);                          // 53,415,936 B
  _Float16* Vh = (_Float16*)(ws + 7630848 + (size_t)53415936);       // 53,415,936 B
  float* aggW  = (float*)(ws + 7630848 + 2*(size_t)53415936);        //    847,872 B
  float* adjsq = (float*)(ws + 7630848 + 2*(size_t)53415936 + 847872);

  k_mlp_recent<<<3726, 256, 0, stream>>>(recent, Wq1, bq1, Wq2, bq2, Qh);
  k_mlp_hist<<<26082, 256, 0, stream>>>(hist, Wk1, bk1, Wk2, bk2,
                                        Wv1, bv1, Wv2, bv2, Kh, Vh);
  k_adjsq<<<169, 32, 0, stream>>>(adj, adjsq);
  k_corr<<<ROWS, 128, 0, stream>>>(Qh, Kh, Vh, corr_w, corr_b,
                                   out + DELAY_OFF, out + TC_OFF, aggW);
  k_gcn<<<64, 256, 0, stream>>>(aggW, adj, adjsq, spQw, spQb, spKw, spKb,
                                spW, gcnW, out + OUT_OFF, out + A_OFF);
}
